// SA_Attn_Mem_83734682403568
// MI455X (gfx1250) — matrix operand order repaired and checked by reading, not compiled, not run on hardware
//
#include <hip/hip_runtime.h>
#include <math.h>

#define BB   8
#define CDIM 128
#define NTOK 4096

typedef __attribute__((ext_vector_type(16))) __bf16 bf16x16;
typedef __attribute__((ext_vector_type(8)))  float  f32x8;
typedef __attribute__((ext_vector_type(4)))  int    v4i;

union FragU {
    bf16x16 f;
    uint4 q[2];
    unsigned int u[8];
};

__device__ __forceinline__ unsigned short f2bf(float x) {
    unsigned int u = __builtin_bit_cast(unsigned int, x);
    u += 0x7FFFu + ((u >> 16) & 1u);   // round-to-nearest-even
    return (unsigned short)(u >> 16);
}

// ---- gfx1250 async global->LDS path (guarded; falls back to sync copy) ----
#if defined(__gfx1250__) && __has_builtin(__builtin_amdgcn_global_load_async_to_lds_b128)
#define HAVE_ASYNC 1
#else
#define HAVE_ASYNC 0
#endif

#if HAVE_ASYNC
typedef __attribute__((address_space(1))) v4i gv4i_t;   // prints as "__device__" in HIP
typedef __attribute__((address_space(3))) v4i lv4i_t;   // prints as "__shared__"
#endif

__device__ __forceinline__ void copy_b128(unsigned short* l, const unsigned short* g) {
#if HAVE_ASYNC
    // AS3 pointers are 32-bit; generic LDS addresses carry the offset in addr[31:0].
    __builtin_amdgcn_global_load_async_to_lds_b128(
        __builtin_bit_cast(gv4i_t*, g),
        __builtin_bit_cast(lv4i_t*, (unsigned int)(unsigned long long)l),
        0, 0);
#else
    *(uint4*)l = *(const uint4*)g;
#endif
}

__device__ __forceinline__ void wait_async0() {
#if HAVE_ASYNC
#if __has_builtin(__builtin_amdgcn_s_wait_asynccnt)
    __builtin_amdgcn_s_wait_asynccnt(0);
#else
    asm volatile("s_wait_asynccnt 0x0" ::: "memory");
#endif
#endif
}

// ---------------------------------------------------------------------------
// Kernel 0: one-time fp32 -> bf16 conversion of the 3 weight matrices.
//   Wb layout: [mat][o][c], 3 * 128 * 128 bf16
// ---------------------------------------------------------------------------
__global__ __launch_bounds__(256) void wcvt_kernel(
    const float* __restrict__ Wq, const float* __restrict__ Wk,
    const float* __restrict__ Wv, unsigned short* __restrict__ Wb)
{
    const int i = blockIdx.x * 256 + threadIdx.x;          // < 3*16384
    const float* src = (i < 16384) ? Wq : ((i < 32768) ? Wk : Wv);
    Wb[i] = f2bf(src[i & 16383]);
}

// ---------------------------------------------------------------------------
// Kernel 1: QKV projection.  Q[b,n,o] = sum_c W[o,c]*x[b,c,n] + bias[o]
//   outputs: Qt [B][N][C] bf16, Kt [B][N][C] bf16, Vc [B][C][N] bf16
// ---------------------------------------------------------------------------
__global__ __launch_bounds__(128) void qkv_proj_kernel(
    const float* __restrict__ h,
    const unsigned short* __restrict__ Wb,   // [3][128][128] bf16
    const float* __restrict__ bq, const float* __restrict__ bk,
    const float* __restrict__ bv,
    unsigned short* __restrict__ Qt,
    unsigned short* __restrict__ Kt,
    unsigned short* __restrict__ Vc)
{
    __shared__ unsigned short xs[64 * CDIM];   // x^T tile [n_local][c], bf16

    const int b    = blockIdx.y;
    const int n0   = blockIdx.x * 64;
    const int t    = threadIdx.x;
    const int lane = t & 31;
    const int wave = t >> 5;
    const int lo   = lane & 15;
    const int hi   = lane >> 4;

    // Stage x^T: h[b][c][n0..n0+63] -> xs[n][c] (coalesced along n)
    {
        const float* xb = h + (size_t)b * CDIM * NTOK;
        const int nn = t & 63;
        const int cb = t >> 6;                 // 0 or 1
        #pragma unroll 8
        for (int i = 0; i < 64; ++i) {
            const int c = cb + 2 * i;
            xs[nn * CDIM + c] = f2bf(xb[(size_t)c * NTOK + n0 + nn]);
        }
    }
    __syncthreads();

    // Per-wave A fragments for its 16 rows
    FragU afr[4];
    {
        const int row = wave * 16 + lo;
        #pragma unroll
        for (int ci = 0; ci < 4; ++ci) {
            const int cbase = ci * 32 + hi * 8;
            afr[ci].q[0] = *(const uint4*)&xs[row * CDIM + cbase];
            afr[ci].q[1] = *(const uint4*)&xs[row * CDIM + cbase + 16];
        }
    }

    const float* bs[3] = { bq, bk, bv };

    #pragma unroll
    for (int mat = 0; mat < 3; ++mat) {
        const unsigned short* Wm = Wb + mat * (CDIM * CDIM);
        for (int ot = 0; ot < CDIM; ot += 16) {
            f32x8 acc;
            #pragma unroll
            for (int r = 0; r < 8; ++r) acc[r] = 0.0f;

            const int o = ot + lo;
            #pragma unroll
            for (int ci = 0; ci < 4; ++ci) {
                // B fragment straight from bf16 weights (two 16B chunks, table k order)
                FragU bf;
                const unsigned short* wp = Wm + o * CDIM + ci * 32 + hi * 8;
                bf.q[0] = *(const uint4*)wp;
                bf.q[1] = *(const uint4*)(wp + 16);
                acc = __builtin_amdgcn_wmma_f32_16x16x32_bf16(
                        false, afr[ci].f, false, bf.f, (short)0, acc, false, false);
            }

            const float bias = bs[mat][o];
            #pragma unroll
            for (int r = 0; r < 8; ++r) {
                const int n = n0 + wave * 16 + r + 8 * hi;
                const unsigned short v = f2bf(acc[r] + bias);
                if (mat == 0)
                    Qt[((size_t)b * NTOK + n) * CDIM + o] = v;
                else if (mat == 1)
                    Kt[((size_t)b * NTOK + n) * CDIM + o] = v;
                else
                    Vc[((size_t)b * CDIM + o) * NTOK + n] = v;
            }
        }
    }
}

// ---------------------------------------------------------------------------
// Kernel 2: flash attention, double-buffered K/V tiles via async-to-LDS.
//   4 waves * 16 query rows; 64-key blocks; online softmax; O += P V^T.
// ---------------------------------------------------------------------------
__device__ __forceinline__ void issue_tile(
    unsigned short* ksb, unsigned short* vsb,
    const unsigned short* Kb, const unsigned short* Vb, int m0, int t)
{
    const int j = t & 7;            // 16-byte chunk index
    const int g = t >> 3;           // 0..15
    #pragma unroll
    for (int i = 0; i < 4; ++i) {   // K tile: [key][c]
        const int m = g + 16 * i;
        copy_b128(&ksb[m * CDIM + j * 8], Kb + (size_t)(m0 + m) * CDIM + j * 8);
    }
    #pragma unroll
    for (int i = 0; i < 8; ++i) {   // V tile: [c][key]
        const int c = g + 16 * i;
        copy_b128(&vsb[c * 64 + j * 8], Vb + (size_t)c * NTOK + m0 + j * 8);
    }
}

__global__ __launch_bounds__(128) void flash_attn_kernel(
    const unsigned short* __restrict__ Qt,   // [B][N][C] bf16
    const unsigned short* __restrict__ Kt,   // [B][N][C] bf16
    const unsigned short* __restrict__ Vc,   // [B][C][N] bf16
    float* __restrict__ out)                 // [B][C][N] f32
{
    __shared__ unsigned short ks[2][64 * CDIM];    // [key][c]
    __shared__ unsigned short vs[2][CDIM * 64];    // [c][key]
    __shared__ unsigned short ps[4][16 * 64];      // per-wave P scratch

    const int b    = blockIdx.y;
    const int n0   = blockIdx.x * 64;
    const int t    = threadIdx.x;
    const int lane = t & 31;
    const int wave = t >> 5;
    const int lo   = lane & 15;
    const int hi   = lane >> 4;

    const unsigned short* Qb = Qt + (size_t)b * NTOK * CDIM;
    const unsigned short* Kb = Kt + (size_t)b * NTOK * CDIM;
    const unsigned short* Vb = Vc + (size_t)b * CDIM * NTOK;

    // Q fragments for this wave's 16 rows (resident in VGPRs)
    FragU qf[4];
    {
        const unsigned short* qr = Qb + (size_t)(n0 + wave * 16 + lo) * CDIM;
        #pragma unroll
        for (int ci = 0; ci < 4; ++ci) {
            const int cbase = ci * 32 + hi * 8;
            qf[ci].q[0] = *(const uint4*)(qr + cbase);
            qf[ci].q[1] = *(const uint4*)(qr + cbase + 16);
        }
    }

    f32x8 oacc[8];
    #pragma unroll
    for (int i = 0; i < 8; ++i)
        #pragma unroll
        for (int r = 0; r < 8; ++r) oacc[i][r] = 0.0f;

    float mrow[8], lrow[8];
    #pragma unroll
    for (int r = 0; r < 8; ++r) { mrow[r] = -__builtin_inff(); lrow[r] = 0.0f; }

    // Prime buffer 0
    issue_tile(ks[0], vs[0], Kb, Vb, 0, t);
    wait_async0();
    __syncthreads();

    for (int it = 0; it < NTOK / 64; ++it) {
        const int cur = it & 1;
        const int m0  = it * 64;

        // Kick off next tile into the other buffer (overlaps with compute)
        if (it + 1 < NTOK / 64)
            issue_tile(ks[1 - cur], vs[1 - cur], Kb, Vb, m0 + 64, t);

        // S = Q * K^T : four 16x16 f32 tiles
        f32x8 s[4];
        #pragma unroll
        for (int jt = 0; jt < 4; ++jt) {
            f32x8 acc;
            #pragma unroll
            for (int r = 0; r < 8; ++r) acc[r] = 0.0f;
            #pragma unroll
            for (int ci = 0; ci < 4; ++ci) {
                FragU kf;
                const unsigned short* kp = &ks[cur][(jt * 16 + lo) * CDIM + ci * 32 + hi * 8];
                kf.q[0] = *(const uint4*)kp;
                kf.q[1] = *(const uint4*)(kp + 16);
                acc = __builtin_amdgcn_wmma_f32_16x16x32_bf16(
                        false, qf[ci].f, false, kf.f, (short)0, acc, false, false);
            }
            s[jt] = acc;
        }

        // Online softmax; rows live in 16-lane halves -> xor-shuffle reduce
        unsigned short* pw = ps[wave];
        #pragma unroll
        for (int r = 0; r < 8; ++r) {
            float v = fmaxf(fmaxf(s[0][r], s[1][r]), fmaxf(s[2][r], s[3][r]));
            v = fmaxf(v, __shfl_xor(v, 1));
            v = fmaxf(v, __shfl_xor(v, 2));
            v = fmaxf(v, __shfl_xor(v, 4));
            v = fmaxf(v, __shfl_xor(v, 8));
            const float mnew = fmaxf(mrow[r], v);

            const int row = r + 8 * hi;
            float rs = 0.0f;
            #pragma unroll
            for (int jt = 0; jt < 4; ++jt) {
                const float p = __expf(s[jt][r] - mnew);
                rs += p;
                pw[row * 64 + jt * 16 + lo] = f2bf(p);
            }
            rs += __shfl_xor(rs, 1);
            rs += __shfl_xor(rs, 2);
            rs += __shfl_xor(rs, 4);
            rs += __shfl_xor(rs, 8);

            const float scale = __expf(mrow[r] - mnew);
            lrow[r] = lrow[r] * scale + rs;
            mrow[r] = mnew;
            #pragma unroll
            for (int i = 0; i < 8; ++i) oacc[i][r] *= scale;
        }

        // DS ops are in-order within a wave; stop compiler reordering only
        asm volatile("" ::: "memory");

        // P back as A fragments (two 16x32 frags over the 64 keys)
        FragU pf[2];
        #pragma unroll
        for (int k0 = 0; k0 < 2; ++k0) {
            const unsigned short* pp = &pw[lo * 64 + k0 * 32 + hi * 8];
            pf[k0].q[0] = *(const uint4*)pp;
            pf[k0].q[1] = *(const uint4*)(pp + 16);
        }

        // O += P * V^T over 8 column tiles of C
        #pragma unroll
        for (int i = 0; i < 8; ++i) {
            #pragma unroll
            for (int k0 = 0; k0 < 2; ++k0) {
                FragU vf;
                const unsigned short* vp = &vs[cur][(i * 16 + lo) * 64 + k0 * 32 + hi * 8];
                vf.q[0] = *(const uint4*)vp;
                vf.q[1] = *(const uint4*)(vp + 16);
                oacc[i] = __builtin_amdgcn_wmma_f32_16x16x32_bf16(
                            false, pf[k0].f, false, vf.f, (short)0, oacc[i], false, false);
            }
        }

        // Next tile must be fully in LDS (and everyone past it) before reuse
        if (it + 1 < NTOK / 64) {
            wait_async0();
            __syncthreads();
        }
    }

    // Epilogue: normalize and store out[b][c][n]
    #pragma unroll
    for (int i = 0; i < 8; ++i) {
        const int c = i * 16 + lo;
        #pragma unroll
        for (int r = 0; r < 8; ++r) {
            const int n = n0 + wave * 16 + r + 8 * hi;
            out[((size_t)b * CDIM + c) * NTOK + n] = oacc[i][r] / lrow[r];
        }
    }
}

// ---------------------------------------------------------------------------
extern "C" void kernel_launch(void* const* d_in, const int* in_sizes, int n_in,
                              void* d_out, int out_size, void* d_ws, size_t ws_size,
                              hipStream_t stream) {
    (void)in_sizes; (void)n_in; (void)out_size; (void)ws_size;
    const float* h  = (const float*)d_in[0];
    const float* Wq = (const float*)d_in[1];
    const float* bq = (const float*)d_in[2];
    const float* Wk = (const float*)d_in[3];
    const float* bk = (const float*)d_in[4];
    const float* Wv = (const float*)d_in[5];
    const float* bv = (const float*)d_in[6];
    float* out = (float*)d_out;

    // bf16 workspace: Qt | Kt | Vc (8 MiB each) | Wb (96 KiB)
    unsigned short* Qt = (unsigned short*)d_ws;
    unsigned short* Kt = Qt + (size_t)BB * NTOK * CDIM;
    unsigned short* Vc = Kt + (size_t)BB * NTOK * CDIM;
    unsigned short* Wb = Vc + (size_t)BB * NTOK * CDIM;

    dim3 grid(NTOK / 64, BB);
    dim3 block(128);
    wcvt_kernel<<<dim3(192), dim3(256), 0, stream>>>(Wq, Wk, Wv, Wb);
    qkv_proj_kernel<<<grid, block, 0, stream>>>(h, Wb, bq, bk, bv, Qt, Kt, Vc);
    flash_attn_kernel<<<grid, block, 0, stream>>>(Qt, Kt, Vc, out);
}
